// GraphResBlock_2224793059399
// MI455X (gfx1250) — compile-verified, matrix operand order checked by reading, not run on hardware
//
#include <hip/hip_runtime.h>
#include <cstdint>

// ---------------- CDNA5 WMMA types ----------------
typedef __attribute__((ext_vector_type(16))) _Float16 v16h;
typedef __attribute__((ext_vector_type(8)))  float    v8f;

#define Nn   65536
#define Cc   256
#define Bb   8
#define EMBD 512
#define NET  7
#define NNT  7
#define GRP  32
#define CPG  8          // Cc / GRP
#define KK   1792       // 7 * 256, K of the main GEMM (multiple of 32)
#define KT   56         // KK / 32
#define EPSn 1e-5f

__device__ __forceinline__ float silu_f(float x) { return x / (1.f + __expf(-x)); }

// ---------------- GroupNorm pass 1: per (batch, channel) sums ----------------
__global__ __launch_bounds__(256) void gn_stats_kernel(
    const float* __restrict__ data, const int* __restrict__ bid,
    float* __restrict__ sum, float* __restrict__ sumsq, float* __restrict__ cnt, int n)
{
    int c  = threadIdx.x;
    int r0 = blockIdx.x * 64;
    if (r0 >= n) return;
    int rend = min(r0 + 64, n);
    int cur = bid[r0];
    int cstart = r0;
    float s = 0.f, q = 0.f;
    for (int r = r0; r < rend; ++r) {
        int b = bid[r];
        if (b != cur) {
            atomicAdd(&sum[cur * Cc + c], s);
            atomicAdd(&sumsq[cur * Cc + c], q);
            if (c == 0) atomicAdd(&cnt[cur], (float)(r - cstart));
            s = 0.f; q = 0.f; cur = b; cstart = r;
        }
        float v = data[(size_t)r * Cc + c];
        s += v; q += v * v;
    }
    atomicAdd(&sum[cur * Cc + c], s);
    atomicAdd(&sumsq[cur * Cc + c], q);
    if (c == 0) atomicAdd(&cnt[cur], (float)(rend - cstart));
}

// ---------------- GroupNorm pass 2: per (batch, group) mean / inv_std ----------------
__global__ void gn_finalize_kernel(
    const float* __restrict__ sum, const float* __restrict__ sumsq,
    const float* __restrict__ cnt, float* __restrict__ gmean, float* __restrict__ ginv)
{
    int t = threadIdx.x;            // 0..255 == Bb * GRP
    int b = t >> 5, g = t & 31;
    float s1 = 0.f, s2 = 0.f;
    for (int j = 0; j < CPG; ++j) {
        int c = g * CPG + j;
        s1 += sum[b * Cc + c];
        s2 += sumsq[b * Cc + c];
    }
    float M    = cnt[b] * (float)CPG;          // count * cpg (reference semantics)
    float inv  = 1.f / (M + EPSn);
    float mean = s1 * inv;
    float var  = (s2 - 2.f * mean * s1 + M * mean * mean) * inv;
    gmean[t] = mean;
    ginv[t]  = rsqrtf(var + EPSn);
}

// ---------------- GroupNorm apply + SiLU -> f16 output ----------------
__global__ __launch_bounds__(256) void gn_apply_silu_kernel(
    const float* __restrict__ data, const int* __restrict__ bid,
    const float* __restrict__ gmean, const float* __restrict__ ginv,
    const float* __restrict__ w, const float* __restrict__ bia, _Float16* __restrict__ out)
{
    size_t idx = (size_t)blockIdx.x * 256 + threadIdx.x;
    int r = (int)(idx >> 8), c = (int)(idx & 255);
    int b = bid[r];
    int g = c >> 3;
    float m  = gmean[b * GRP + g];
    float iv = ginv [b * GRP + g];
    float y  = (data[idx] - m) * iv * w[c] + bia[c];
    out[idx] = (_Float16)silu_f(y);
}

// ---------------- Edge scatter (f16): scat[(row*7+et), :] += h[col, :] ----------------
// Uses CDNA5 GLOBAL_ATOMIC_PK_ADD_F16 (packed 2xf16 atomic add, no return).
__global__ __launch_bounds__(128) void scatter_f16_kernel(
    const unsigned* __restrict__ h2 /* [N,128] f16x2 */,
    const int* __restrict__ erow, const int* __restrict__ ecol,
    const int* __restrict__ etype, const int* __restrict__ ntype,
    unsigned* __restrict__ scat2 /* [N*7,128] f16x2 */,
    float* __restrict__ cntoh, int nE)
{
    int c  = threadIdx.x;          // 0..127 : packed channel pair
    int e0 = blockIdx.x * 8;
    for (int j = 0; j < 8; ++j) {
        int e = e0 + j;
        if (e >= nE) break;
        int rr = erow[e], cc = ecol[e], t = etype[e];
        if (j < 7 && e + 1 < nE) {
            // warm L2 for the next edge's gather row (global_prefetch_b8)
            __builtin_prefetch(h2 + (size_t)ecol[e + 1] * 128 + c, 0, 1);
        }
        unsigned val = h2[(size_t)cc * 128 + c];     // h is L2-resident (32MB)
        unsigned long long addr =
            (unsigned long long)(scat2 + ((size_t)rr * NET + t) * 128 + c);
        asm volatile("global_atomic_pk_add_f16 %0, %1, off"
                     :: "v"(addr), "v"(val) : "memory");
        if (c == 0) atomicAdd(&cntoh[((size_t)rr * NET + t) * 8 + ntype[cc]], 1.0f);
    }
}

// ---------------- Pack weights into CDNA5 B-fragment lane layout ----------------
// B frag (16x16x32, 16-bit): lane 0..15 -> n=lane, K = 0..7 & 16..23;
// lane 16..31 -> n=lane-16, K = 8..15 & 24..31 (half i<8 -> K+=i, i>=8 -> K+=8+i).
__global__ __launch_bounds__(256) void pack_w_kernel(
    const float* __restrict__ w /* [7*263, 256] */, _Float16* __restrict__ wp)
{
    int idx  = blockIdx.x * 256 + threadIdx.x;   // < KK*256
    int i    = idx & 15;
    int lane = (idx >> 4) & 31;
    int nf   = (idx >> 9) & 15;
    int kt   = idx >> 13;
    int k  = kt * 32 + ((lane < 16) ? 0 : 8) + ((i < 8) ? i : 8 + i);
    int nn = nf * 16 + (lane & 15);
    int t  = k >> 8, ch = k & 255;               // main K skips one-hot rows
    wp[idx] = (_Float16)w[(size_t)(t * 263 + ch) * Cc + nn];
}

// ---------------- WMMA GEMM: C[N,256] = A[N,1792](f16) @ Wpack ----------------
// A tiles staged via CDNA5 async DMA (GLOBAL_LOAD_ASYNC_TO_LDS_B128, ASYNCcnt),
// double-buffered in LDS so the DMA for tile kt+1 overlaps the WMMAs of tile kt.
__global__ __launch_bounds__(256) void gemm_wmma_kernel(
    const _Float16* __restrict__ A, const _Float16* __restrict__ Bp, float* __restrict__ Cm)
{
    __shared__ __align__(16) _Float16 As[2][64 * 32];   // 2 x 4KB A tiles (f16)
    int tid  = threadIdx.x;
    int lane = tid & 31, wave = tid >> 5;
    int wm = wave >> 2;          // 0..1 -> 32-row slab
    int wn = wave & 3;           // 0..3 -> 64-col slab
    int m0 = blockIdx.x * 64;

    v8f acc[2][4];
    v8f z = {};
    for (int a = 0; a < 2; ++a)
        for (int b = 0; b < 4; ++b) acc[a][b] = z;

    int lrow = tid >> 2;          // 0..63 : A-tile row this thread stages
    int lseg = (tid & 3) * 8;     // 0,8,16,24 : 8-half (16B) segment
    int mrow = lane & 15;
    int koff = (lane < 16) ? 0 : 8;

    // wave-relative LDS byte offsets (low 32 bits of the flat shared address)
    unsigned lds_st0 = (unsigned)(unsigned long long)&As[0][lrow * 32 + lseg];
    unsigned lds_st1 = (unsigned)(unsigned long long)&As[1][lrow * 32 + lseg];
    const _Float16* ga_base = A + (size_t)(m0 + lrow) * KK + lseg;

    union F16 { v16h v; float4 f4[2]; };

    // prologue: DMA tile 0 into buffer 0
    {
        unsigned long long ga = (unsigned long long)ga_base;
        asm volatile("global_load_async_to_lds_b128 %0, %1, off"
                     :: "v"(lds_st0), "v"(ga) : "memory");
        asm volatile("s_wait_asynccnt 0" ::: "memory");
    }
    __syncthreads();

    for (int kt = 0; kt < KT; ++kt) {
        int cur = kt & 1;
        // kick off async DMA for the next tile into the other buffer
        if (kt + 1 < KT) {
            unsigned long long ga = (unsigned long long)(ga_base + (size_t)(kt + 1) * 32);
            unsigned ldst = (cur == 0) ? lds_st1 : lds_st0;
            asm volatile("global_load_async_to_lds_b128 %0, %1, off"
                         :: "v"(ldst), "v"(ga) : "memory");
        }

        // A fragments from LDS (ISA §7.12.2 16-bit A layout)
        F16 af[2];
        for (int fm = 0; fm < 2; ++fm) {
            const float4* lp = reinterpret_cast<const float4*>(
                &As[cur][(wm * 32 + fm * 16 + mrow) * 32 + koff]);
            af[fm].f4[0] = lp[0];     // K koff..koff+7
            af[fm].f4[1] = lp[2];     // K koff+16..koff+23
        }
        // B fragments straight from pre-packed (L2-resident) weights
        for (int fn = 0; fn < 4; ++fn) {
            int nf = wn * 4 + fn;
            const float4* bp = reinterpret_cast<const float4*>(
                Bp + (((size_t)kt * 16 + nf) * 32 + lane) * 16);
            F16 bf; bf.f4[0] = bp[0]; bf.f4[1] = bp[1];
            for (int fm = 0; fm < 2; ++fm) {
                acc[fm][fn] = __builtin_amdgcn_wmma_f32_16x16x32_f16(
                    false, af[fm].v, false, bf.v, (short)0, acc[fm][fn], false, false);
            }
        }

        // drain this wave's DMA, then block barrier before consuming next tile
        if (kt + 1 < KT) {
            asm volatile("s_wait_asynccnt 0" ::: "memory");
            __syncthreads();
        }
    }

    // epilogue: 32-bit C/D layout (VGPR r -> M=r / M=8+r for lane hi half)
    int mb  = m0 + wm * 32 + ((lane < 16) ? 0 : 8);
    int nc0 = wn * 64 + (lane & 15);
    for (int fm = 0; fm < 2; ++fm)
        for (int fn = 0; fn < 4; ++fn) {
            int nn = nc0 + fn * 16;
            for (int r = 0; r < 8; ++r)
                Cm[(size_t)(mb + fm * 16 + r) * Cc + nn] = acc[fm][fn][r];
        }
}

// ---------------- emb_out = silu(emb) @ emb_w + emb_b ----------------
__global__ __launch_bounds__(256) void emb_kernel(
    const float* __restrict__ emb, const float* __restrict__ ew,
    const float* __restrict__ eb, float* __restrict__ out)
{
    int b = blockIdx.x, c = threadIdx.x;
    float acc = eb[c];
    for (int k = 0; k < EMBD; ++k)
        acc += silu_f(emb[b * EMBD + k]) * ew[(size_t)k * Cc + c];
    out[b * Cc + c] = acc;
}

// ---------------- conv1 epilogue: + one-hot correction + emb_out[batch] ----------------
__global__ __launch_bounds__(256) void post1_kernel(
    const float* __restrict__ conv, const float* __restrict__ cntoh,
    const float* __restrict__ w, const float* __restrict__ embout,
    const int* __restrict__ bid, float* __restrict__ out)
{
    size_t idx = (size_t)blockIdx.x * 256 + threadIdx.x;
    int r = (int)(idx >> 8), c = (int)(idx & 255);
    float corr = 0.f;
    for (int t = 0; t < NET; ++t) {
        const float* cp = cntoh + ((size_t)r * NET + t) * 8;
        const float* wr = w + (size_t)(t * 263 + 256) * Cc + c;
        for (int j = 0; j < NNT; ++j) {
            float cv = cp[j];
            if (cv != 0.f) corr += cv * wr[(size_t)j * Cc];
        }
    }
    out[idx] = conv[idx] + corr + embout[bid[r] * Cc + c];
}

// ---------------- conv2 epilogue + residual ----------------
__global__ __launch_bounds__(256) void final_kernel(
    const float* __restrict__ x, const float* __restrict__ conv,
    const float* __restrict__ cntoh, const float* __restrict__ w, float* __restrict__ out)
{
    size_t idx = (size_t)blockIdx.x * 256 + threadIdx.x;
    int r = (int)(idx >> 8), c = (int)(idx & 255);
    float corr = 0.f;
    for (int t = 0; t < NET; ++t) {
        const float* cp = cntoh + ((size_t)r * NET + t) * 8;
        const float* wr = w + (size_t)(t * 263 + 256) * Cc + c;
        for (int j = 0; j < NNT; ++j) {
            float cv = cp[j];
            if (cv != 0.f) corr += cv * wr[(size_t)j * Cc];
        }
    }
    out[idx] = x[idx] + conv[idx] + corr;
}

// =====================================================================
extern "C" void kernel_launch(void* const* d_in, const int* in_sizes, int n_in,
                              void* d_out, int out_size, void* d_ws, size_t ws_size,
                              hipStream_t stream)
{
    const float* x     = (const float*)d_in[0];
    const float* emb   = (const float*)d_in[1];
    const int*   bid   = (const int*)  d_in[2];
    const int*   eidx  = (const int*)  d_in[3];
    const int*   etype = (const int*)  d_in[4];
    const int*   ntype = (const int*)  d_in[5];
    const float* gn1w  = (const float*)d_in[6];
    const float* gn1b  = (const float*)d_in[7];
    const float* w1    = (const float*)d_in[8];
    const float* embw  = (const float*)d_in[9];
    const float* embb  = (const float*)d_in[10];
    const float* gn2w  = (const float*)d_in[11];
    const float* gn2b  = (const float*)d_in[12];
    const float* w2    = (const float*)d_in[13];
    float* out = (float*)d_out;

    const int nE = in_sizes[3] / 2;
    const int* erow = eidx;
    const int* ecol = eidx + nE;

    // ---- workspace layout ----
    char*  ws  = (char*)d_ws;
    size_t off = 0;
    auto take = [&](size_t bytes) -> char* {
        char* p = ws + off;
        off = (off + bytes + 255) & ~(size_t)255;
        return p;
    };
    _Float16* scat16 = (_Float16*)take((size_t)Nn * NET * Cc * 2);  // 235 MB (f16)
    float*    cntoh  = (float*)   take((size_t)Nn * NET * 8  * 4);  // 14.7 MB
    _Float16* hbuf   = (_Float16*)take((size_t)Nn * Cc * 2);        // 32 MB (f16)
    float*    pbuf   = (float*)   take((size_t)Nn * Cc * 4);        // 64 MB
    float*    cbuf   = (float*)   take((size_t)Nn * Cc * 4);        // 64 MB
    _Float16* wp1    = (_Float16*)take((size_t)KK * Cc * 2);        // 0.9 MB
    _Float16* wp2    = (_Float16*)take((size_t)KK * Cc * 2);        // 0.9 MB
    float*    stats  = (float*)   take(2 * Bb * Cc * 4 + 256);      // sum|sumsq|cnt
    float*    sum    = stats;
    float*    sumsq  = stats + Bb * Cc;
    float*    cnt    = stats + 2 * Bb * Cc;
    float*    gmean  = (float*)   take(Bb * GRP * 4);
    float*    ginv   = (float*)   take(Bb * GRP * 4);
    float*    embout = (float*)   take(Bb * Cc * 4);

    const size_t scat_bytes  = (size_t)Nn * NET * Cc * 2;
    const size_t cntoh_bytes = (size_t)Nn * NET * 8  * 4;
    const size_t stats_bytes = 2 * Bb * Cc * 4 + 256;

    unsigned* h2    = (unsigned*)hbuf;
    unsigned* scat2 = (unsigned*)scat16;

    // pack both weight matrices (cheap, overlaps with GN work)
    pack_w_kernel<<<KK * Cc / 256, 256, 0, stream>>>(w1, wp1);
    pack_w_kernel<<<KK * Cc / 256, 256, 0, stream>>>(w2, wp2);
    emb_kernel<<<Bb, 256, 0, stream>>>(emb, embw, embb, embout);

    // ---------- block 1: GN1 + SiLU -> h (f16) ----------
    hipMemsetAsync(stats, 0, stats_bytes, stream);
    gn_stats_kernel<<<Nn / 64, 256, 0, stream>>>(x, bid, sum, sumsq, cnt, Nn);
    gn_finalize_kernel<<<1, 256, 0, stream>>>(sum, sumsq, cnt, gmean, ginv);
    gn_apply_silu_kernel<<<(Nn * Cc) / 256, 256, 0, stream>>>(x, bid, gmean, ginv, gn1w, gn1b, hbuf);

    // ---------- conv1 ----------
    hipMemsetAsync(scat16, 0, scat_bytes, stream);
    hipMemsetAsync(cntoh, 0, cntoh_bytes, stream);
    scatter_f16_kernel<<<(nE + 7) / 8, 128, 0, stream>>>(h2, erow, ecol, etype, ntype, scat2, cntoh, nE);
    gemm_wmma_kernel<<<Nn / 64, 256, 0, stream>>>(scat16, wp1, cbuf);
    post1_kernel<<<(Nn * Cc) / 256, 256, 0, stream>>>(cbuf, cntoh, w1, embout, bid, pbuf);

    // ---------- block 2: GN2 + SiLU -> h (f16) ----------
    hipMemsetAsync(stats, 0, stats_bytes, stream);
    gn_stats_kernel<<<Nn / 64, 256, 0, stream>>>(pbuf, bid, sum, sumsq, cnt, Nn);
    gn_finalize_kernel<<<1, 256, 0, stream>>>(sum, sumsq, cnt, gmean, ginv);
    gn_apply_silu_kernel<<<(Nn * Cc) / 256, 256, 0, stream>>>(pbuf, bid, gmean, ginv, gn2w, gn2b, hbuf);

    // ---------- conv2 + residual ----------
    hipMemsetAsync(scat16, 0, scat_bytes, stream);
    hipMemsetAsync(cntoh, 0, cntoh_bytes, stream);
    scatter_f16_kernel<<<(nE + 7) / 8, 128, 0, stream>>>(h2, erow, ecol, etype, ntype, scat2, cntoh, nE);
    gemm_wmma_kernel<<<Nn / 64, 256, 0, stream>>>(scat16, wp2, cbuf);
    final_kernel<<<(Nn * Cc) / 256, 256, 0, stream>>>(x, cbuf, cntoh, w2, out);

    (void)ws_size; (void)n_in; (void)out_size;
}
